// MambaBlock_27797028340369
// MI455X (gfx1250) — matrix operand order repaired and checked by reading, not compiled, not run on hardware
//
#include <hip/hip_runtime.h>
#include <hip/hip_bf16.h>

#define D_MODEL 1024
#define D_STATE 16
#define D_CONV  4
#define D_INNER 2048
#define DT_RANK 64
#define SEQ     1024
#define BATCH   2
#define NTOK    (BATCH * SEQ)            // 2048 tokens
#define NXZ     (2 * D_INNER)            // 4096
#define NDBL    (DT_RANK + 2 * D_STATE)  // 96

typedef _Float16 h8  __attribute__((ext_vector_type(8)));
typedef _Float16 h16 __attribute__((ext_vector_type(16)));
typedef float    f8  __attribute__((ext_vector_type(8)));

// ---------------------------------------------------------------------------
// CDNA5 async global->LDS copy (ASYNCcnt path), with compile-safe fallback.
// Builtin signature (from hipcc diagnostic): param0 = int vector_size(16) in
// AS1 (__device__), param1 = LDS side, then imm offset + cpol.
// ---------------------------------------------------------------------------
#if __has_builtin(__builtin_amdgcn_global_load_async_to_lds_b128)
#define HAVE_ASYNC_LDS 1
typedef int v4i_vs __attribute__((vector_size(16)));
typedef __attribute__((address_space(1))) v4i_vs glb_v4i;
typedef __attribute__((address_space(3))) v4i_vs lds_v4i;
__device__ __forceinline__ void async_b128(const _Float16* g, _Float16* l) {
    __builtin_amdgcn_global_load_async_to_lds_b128(
        (glb_v4i*)g, (lds_v4i*)l, 0, 0);
}
#else
#define HAVE_ASYNC_LDS 0
#endif

__device__ __forceinline__ void stage_wait() {
#if HAVE_ASYNC_LDS
#if __has_builtin(__builtin_amdgcn_s_wait_asynccnt)
    __builtin_amdgcn_s_wait_asynccnt(0);
#else
    asm volatile("s_wait_asynccnt 0x0" ::: "memory");
#endif
#endif
}

// Stage one K-step of the 64x128 block tile into LDS.
// A tile: 64 rows x 32 halves (one b128 chunk per thread).
// B tile: 128 rows x 32 halves (two b128 chunks per thread).
__device__ __forceinline__ void stage_tile(const _Float16* __restrict__ A,
                                           const _Float16* __restrict__ B,
                                           _Float16* sA, _Float16* sB,
                                           int bm, int bn, int K, int k0, int tid) {
    const int arow = tid >> 2, akc = tid & 3;
    const _Float16* ag = A + (size_t)(bm * 64 + arow) * K + k0 + akc * 8;
    _Float16* al = sA + arow * 32 + akc * 8;
    const int j1 = tid + 256;
    const int b0row = tid >> 2, b0kc = tid & 3;
    const int b1row = j1 >> 2,  b1kc = j1 & 3;
    const _Float16* bg0 = B + (size_t)(bn * 128 + b0row) * K + k0 + b0kc * 8;
    const _Float16* bg1 = B + (size_t)(bn * 128 + b1row) * K + k0 + b1kc * 8;
    _Float16* bl0 = sB + b0row * 32 + b0kc * 8;
    _Float16* bl1 = sB + b1row * 32 + b1kc * 8;
#if HAVE_ASYNC_LDS
    async_b128(ag, al);
    async_b128(bg0, bl0);
    async_b128(bg1, bl1);
#else
    h8 a = *(const h8*)ag, b0 = *(const h8*)bg0, b1 = *(const h8*)bg1;
    *(h8*)al = a; *(h8*)bl0 = b0; *(h8*)bl1 = b1;
#endif
}

// ---------------------------------------------------------------------------
// LDS-staged double-buffered WMMA GEMM: C[M,N] = A[M,K] * B[N,K]^T
// Block: 256 threads = 8 waves in a 2(M) x 4(N) grid; wave tile 32x32
// (MT=2 x NT=2 wmma accumulators); block tile 64x128; K-step 32.
// Requires M%64==0, N%128==0, K%32==0.
// ---------------------------------------------------------------------------
template <int ADD_RESID>
__global__ __launch_bounds__(256)
void wmma_gemm_lds_kernel(const _Float16* __restrict__ A,
                          const _Float16* __restrict__ B,
                          float* __restrict__ C,
                          const float* __restrict__ resid,
                          int M, int N, int K) {
    __shared__ _Float16 sA[2][64 * 32];   // 2 x 4KB
    __shared__ _Float16 sB[2][128 * 32];  // 2 x 8KB
    const int tid  = threadIdx.x;
    const int lane = tid & 31;
    const int wid  = tid >> 5;
    const int wr   = wid & 1;       // wave row   (2)
    const int wc   = wid >> 1;      // wave col   (4)
    const int half = lane >> 4;
    const int l15  = lane & 15;
    const int bn   = blockIdx.x;
    const int bm   = blockIdx.y;

    const f8 fzero = {0.f, 0.f, 0.f, 0.f, 0.f, 0.f, 0.f, 0.f};
    f8 acc[2][2];
#pragma unroll
    for (int mt = 0; mt < 2; ++mt)
#pragma unroll
        for (int nt = 0; nt < 2; ++nt) acc[mt][nt] = fzero;

    stage_tile(A, B, sA[0], sB[0], bm, bn, K, 0, tid);
    stage_wait();
    __syncthreads();

    int cur = 0;
    for (int k0 = 0; k0 < K; k0 += 32) {
        const bool has_next = (k0 + 32) < K;
        if (has_next)  // issue next stage before compute so it overlaps 4 wmmas
            stage_tile(A, B, sA[cur ^ 1], sB[cur ^ 1], bm, bn, K, k0 + 32, tid);

        // A fragments (lane m=l15, contiguous K chunk [half*16,+16))
        h16 afrag[2];
#pragma unroll
        for (int mt = 0; mt < 2; ++mt) {
            const _Float16* ap = &sA[cur][(wr * 32 + mt * 16 + l15) * 32];
            h8 lo = *(const h8*)(ap + half * 16);
            h8 hi = *(const h8*)(ap + half * 16 + 8);
#pragma unroll
            for (int i = 0; i < 8; ++i) { afrag[mt][i] = lo[i]; afrag[mt][8 + i] = hi[i]; }
        }
#pragma unroll
        for (int nt = 0; nt < 2; ++nt) {
            // B fragment: column n=l15, contiguous K chunk [half*16,+16)
            h16 b = *(const h16*)(&sB[cur][(wc * 32 + nt * 16 + l15) * 32 + half * 16]);
#pragma unroll
            for (int mt = 0; mt < 2; ++mt)
                acc[mt][nt] = __builtin_amdgcn_wmma_f32_16x16x32_f16(
                    false, afrag[mt], false, b, (short)0, acc[mt][nt], false, false);
        }
        if (has_next) {
            stage_wait();
            __syncthreads();
            cur ^= 1;
        }
    }

    // D layout: vgpr r, lane -> (m = r + 8*half, n = l15)
#pragma unroll
    for (int mt = 0; mt < 2; ++mt)
#pragma unroll
        for (int nt = 0; nt < 2; ++nt) {
            const int ncol = bn * 128 + wc * 32 + nt * 16 + l15;
#pragma unroll
            for (int r = 0; r < 8; ++r) {
                const int mrow = bm * 64 + wr * 32 + mt * 16 + half * 8 + r;
                const size_t idx = (size_t)mrow * N + ncol;
                float v = acc[mt][nt][r];
                if (ADD_RESID) v += resid[idx];
                C[idx] = v;
            }
        }
}

// ---------------------------------------------------------------------------
// Direct (L2-reuse) WMMA GEMM for skinny N (x_proj, N=96). Wave tile 16x32.
// ---------------------------------------------------------------------------
__global__ void wmma_gemm_direct_kernel(const _Float16* __restrict__ A,
                                        const _Float16* __restrict__ B,
                                        float* __restrict__ C,
                                        int M, int N, int K) {
    const int lane = threadIdx.x & 31;
    const int gw   = blockIdx.x * (blockDim.x >> 5) + (threadIdx.x >> 5);
    const int tiles_ng = N / 32;
    if (gw >= (M >> 4) * tiles_ng) return;  // wave-uniform exit
    const int tm = gw / tiles_ng;
    const int tn = gw - tm * tiles_ng;
    const int half = lane >> 4, l15 = lane & 15;

    const _Float16* arow = A + (size_t)(tm * 16 + l15) * K;
    const f8 fzero = {0.f, 0.f, 0.f, 0.f, 0.f, 0.f, 0.f, 0.f};
    f8 acc0 = fzero, acc1 = fzero;

    for (int k0 = 0; k0 < K; k0 += 32) {
        if (k0 + 64 < K) __builtin_prefetch((const void*)(arow + k0 + 64), 0, 3);
        h8 alo = *(const h8*)(arow + k0 + half * 16);
        h8 ahi = *(const h8*)(arow + k0 + half * 16 + 8);
        h16 a;
#pragma unroll
        for (int i = 0; i < 8; ++i) { a[i] = alo[i]; a[8 + i] = ahi[i]; }
        h16 b0 = *(const h16*)(B + (size_t)(tn * 32 + l15) * K + k0 + half * 16);
        h16 b1 = *(const h16*)(B + (size_t)(tn * 32 + 16 + l15) * K + k0 + half * 16);
        acc0 = __builtin_amdgcn_wmma_f32_16x16x32_f16(false, a, false, b0, (short)0, acc0, false, false);
        acc1 = __builtin_amdgcn_wmma_f32_16x16x32_f16(false, a, false, b1, (short)0, acc1, false, false);
    }
#pragma unroll
    for (int r = 0; r < 8; ++r) {
        const int mrow = tm * 16 + half * 8 + r;
        C[(size_t)mrow * N + tn * 32 + l15]      = acc0[r];
        C[(size_t)mrow * N + tn * 32 + 16 + l15] = acc1[r];
    }
}

// ---------------------------------------------------------------------------
// f32 -> f16 converts
// ---------------------------------------------------------------------------
__global__ void cvt_f32_f16_kernel(const float* __restrict__ src,
                                   _Float16* __restrict__ dst, int n) {
    int i = blockIdx.x * 256 + threadIdx.x;
    if (i < n) dst[i] = (_Float16)src[i];
}

__global__ void extract_cvt_kernel(const float* __restrict__ src,
                                   _Float16* __restrict__ dst,
                                   int rows, int stride, int col0, int cols) {
    int i = blockIdx.x * 256 + threadIdx.x;
    if (i >= rows * cols) return;
    int r = i / cols, c = i - r * cols;
    dst[i] = (_Float16)src[(size_t)r * stride + col0 + c];
}

// ---------------------------------------------------------------------------
// RMSNorm -> f16. One block (256 thr) per token.
// ---------------------------------------------------------------------------
__global__ void rmsnorm_f16_kernel(const float* __restrict__ x,
                                   const float* __restrict__ w,
                                   _Float16* __restrict__ h) {
    __shared__ float red[256];
    const int t = blockIdx.x;
    const float* xr = x + (size_t)t * D_MODEL;
    float ss = 0.f;
    for (int i = threadIdx.x; i < D_MODEL; i += 256) { float v = xr[i]; ss += v * v; }
    red[threadIdx.x] = ss;
    __syncthreads();
    for (int s = 128; s > 0; s >>= 1) {
        if (threadIdx.x < s) red[threadIdx.x] += red[threadIdx.x + s];
        __syncthreads();
    }
    const float scale = rsqrtf(red[0] * (1.0f / D_MODEL) + 1e-5f);
    for (int i = threadIdx.x; i < D_MODEL; i += 256)
        h[(size_t)t * D_MODEL + i] = (_Float16)(xr[i] * scale * w[i]);
}

// ---------------------------------------------------------------------------
// Depthwise causal conv (4) + SiLU
// ---------------------------------------------------------------------------
__global__ void conv_silu_kernel(const float* __restrict__ xz,
                                 const float* __restrict__ cw,
                                 const float* __restrict__ cb,
                                 float* __restrict__ xs32,
                                 _Float16* __restrict__ xs16) {
    int i = blockIdx.x * 256 + threadIdx.x;
    if (i >= NTOK * D_INNER) return;
    const int c = i & (D_INNER - 1);
    const int t = i >> 11;
    const int l = t & (SEQ - 1);
    const int bbase = t - l;
    float acc = cb[c];
#pragma unroll
    for (int k = 0; k < D_CONV; ++k) {
        int ls = l - (D_CONV - 1) + k;
        if (ls >= 0) acc += cw[c * D_CONV + k] * xz[(size_t)(bbase + ls) * NXZ + c];
    }
    const float s = acc / (1.f + __expf(-acc));
    xs32[i] = s;
    xs16[i] = (_Float16)s;
}

// ---------------------------------------------------------------------------
// Selective scan: lane&15 = state n; 2 channels per wave; unrolled x4 so 16
// independent loads are in flight ahead of the dependent state recurrence.
// ---------------------------------------------------------------------------
__global__ void scan_kernel(const float* __restrict__ dt_lin,
                            const float* __restrict__ xs32,
                            const float* __restrict__ xdbl,
                            const float* __restrict__ xz,
                            const float* __restrict__ dt_b,
                            const float* __restrict__ A_log,
                            const float* __restrict__ Dvec,
                            _Float16* __restrict__ y16) {
    const int g    = blockIdx.x * 256 + threadIdx.x;
    const int lane = threadIdx.x & 31;
    const int n    = lane & 15;
    const int pair = lane >> 4;
    const int bc   = (g >> 5) * 2 + pair;
    const int b    = bc >> 11;
    const int c    = bc & (D_INNER - 1);
    const int tb   = b << 10;

    const float An  = -__expf(A_log[c * D_STATE + n]);
    const float Dc  = Dvec[c];
    const float dtb = dt_b[c];
    float hst = 0.f;

    for (int l = 0; l < SEQ; l += 4) {
        float dl[4], xv[4], Bn[4], Cn[4];
#pragma unroll
        for (int u = 0; u < 4; ++u) {
            const int t = tb + l + u;
            dl[u] = dt_lin[(size_t)t * D_INNER + c];
            xv[u] = xs32[(size_t)t * D_INNER + c];
            Bn[u] = xdbl[t * NDBL + DT_RANK + n];
            Cn[u] = xdbl[t * NDBL + DT_RANK + D_STATE + n];
        }
#pragma unroll
        for (int u = 0; u < 4; ++u) {
            const float d0 = dl[u] + dtb;
            const float dt = (d0 > 20.f) ? d0 : log1pf(__expf(d0));
            hst = __expf(dt * An) * hst + dt * xv[u] * Bn[u];
            float p = hst * Cn[u];
            p += __shfl_xor(p, 8);
            p += __shfl_xor(p, 4);
            p += __shfl_xor(p, 2);
            p += __shfl_xor(p, 1);
            if (n == 0) {
                const int t = tb + l + u;
                float y = p + Dc * xv[u];
                const float z = xz[(size_t)t * NXZ + D_INNER + c];
                y *= z / (1.f + __expf(-z));
                y16[(size_t)t * D_INNER + c] = (_Float16)y;
            }
        }
    }
}

// ---------------------------------------------------------------------------
extern "C" void kernel_launch(void* const* d_in, const int* in_sizes, int n_in,
                              void* d_out, int out_size, void* d_ws, size_t ws_size,
                              hipStream_t stream) {
    const float* x      = (const float*)d_in[0];
    const float* norm_w = (const float*)d_in[1];
    const float* in_w   = (const float*)d_in[2];
    const float* conv_w = (const float*)d_in[3];
    const float* conv_b = (const float*)d_in[4];
    const float* xp_w   = (const float*)d_in[5];
    const float* dt_w   = (const float*)d_in[6];
    const float* dt_b   = (const float*)d_in[7];
    const float* A_log  = (const float*)d_in[8];
    const float* Dvec   = (const float*)d_in[9];
    const float* out_w  = (const float*)d_in[10];
    float* out          = (float*)d_out;

    size_t off = 0;
    auto carve = [&](size_t bytes) -> void* {
        void* p = (char*)d_ws + off;
        off += (bytes + 255) & ~(size_t)255;
        return p;
    };
    _Float16* h16v   = (_Float16*)carve((size_t)NTOK * D_MODEL * 2);
    _Float16* Win16  = (_Float16*)carve((size_t)NXZ * D_MODEL * 2);
    _Float16* Wxp16  = (_Float16*)carve((size_t)NDBL * D_INNER * 2);
    _Float16* Wdt16  = (_Float16*)carve((size_t)D_INNER * DT_RANK * 2);
    _Float16* Wout16 = (_Float16*)carve((size_t)D_MODEL * D_INNER * 2);
    float*    xzbuf  = (float*)carve((size_t)NTOK * NXZ * 4);
    _Float16* xs16   = (_Float16*)carve((size_t)NTOK * D_INNER * 2);
    float*    xs32   = (float*)carve((size_t)NTOK * D_INNER * 4);
    float*    xdbl   = (float*)carve((size_t)NTOK * NDBL * 4);
    _Float16* dtr16  = (_Float16*)carve((size_t)NTOK * DT_RANK * 2);
    float*    dtlin  = (float*)carve((size_t)NTOK * D_INNER * 4);
    _Float16* y16    = (_Float16*)carve((size_t)NTOK * D_INNER * 2);

    // 1. weight converts
    cvt_f32_f16_kernel<<<(NXZ * D_MODEL + 255) / 256, 256, 0, stream>>>(in_w, Win16, NXZ * D_MODEL);
    cvt_f32_f16_kernel<<<(NDBL * D_INNER + 255) / 256, 256, 0, stream>>>(xp_w, Wxp16, NDBL * D_INNER);
    cvt_f32_f16_kernel<<<(D_INNER * DT_RANK + 255) / 256, 256, 0, stream>>>(dt_w, Wdt16, D_INNER * DT_RANK);
    cvt_f32_f16_kernel<<<(D_MODEL * D_INNER + 255) / 256, 256, 0, stream>>>(out_w, Wout16, D_MODEL * D_INNER);

    // 2. RMSNorm
    rmsnorm_f16_kernel<<<NTOK, 256, 0, stream>>>(x, norm_w, h16v);

    // 3. in_proj: [2048,4096] = h[2048,1024] x Win[4096,1024]^T
    wmma_gemm_lds_kernel<0><<<dim3(NXZ / 128, NTOK / 64), 256, 0, stream>>>(
        h16v, Win16, xzbuf, nullptr, NTOK, NXZ, D_MODEL);

    // 4. conv + SiLU
    conv_silu_kernel<<<(NTOK * D_INNER + 255) / 256, 256, 0, stream>>>(
        xzbuf, conv_w, conv_b, xs32, xs16);

    // 5. x_proj: [2048,96] = xs[2048,2048] x Wxp[96,2048]^T (skinny N -> direct)
    {
        int waves = (NTOK / 16) * (NDBL / 32);
        wmma_gemm_direct_kernel<<<(waves + 3) / 4, 128, 0, stream>>>(
            xs16, Wxp16, xdbl, NTOK, NDBL, D_INNER);
    }

    // 6. dt_r -> f16, dt GEMM [2048,2048], K=64
    extract_cvt_kernel<<<(NTOK * DT_RANK + 255) / 256, 256, 0, stream>>>(
        xdbl, dtr16, NTOK, NDBL, 0, DT_RANK);
    wmma_gemm_lds_kernel<0><<<dim3(D_INNER / 128, NTOK / 64), 256, 0, stream>>>(
        dtr16, Wdt16, dtlin, nullptr, NTOK, D_INNER, DT_RANK);

    // 7. selective scan + gate -> y16
    scan_kernel<<<(BATCH * D_INNER * D_STATE) / 256, 256, 0, stream>>>(
        dtlin, xs32, xdbl, xzbuf, dt_b, A_log, Dvec, y16);

    // 8. out_proj + residual
    wmma_gemm_lds_kernel<1><<<dim3(D_MODEL / 128, NTOK / 64), 256, 0, stream>>>(
        y16, Wout16, out, x, NTOK, D_MODEL, D_INNER);

    (void)in_sizes; (void)n_in; (void)out_size; (void)ws_size;
}
